// Multi_Head_Attention_20066087207357
// MI455X (gfx1250) — matrix operand order repaired and checked by reading, not compiled, not run on hardware
//
#include <hip/hip_runtime.h>
#include <hip/hip_bf16.h>

// ---------------------------------------------------------------------------
// MHA block for MI455X (gfx1250): bf16 WMMA GEMMs (async-LDS double-buffered)
// + flash-attention + LayerNorm.  Shapes: B=2, T=2048, H=1024, 16 heads, d=64.
// ---------------------------------------------------------------------------

typedef __attribute__((ext_vector_type(16))) __bf16    v16bf;
typedef __attribute__((ext_vector_type(8)))  float     v8f;
typedef __attribute__((ext_vector_type(4)))  unsigned  u32x4;

union BFrag {
    v16bf v;
    u32x4 q[2];
    unsigned short u[16];
};

__device__ __forceinline__ unsigned short f2bf(float x) {
    union { float f; unsigned u; } c; c.f = x;
    unsigned u = c.u + 0x7FFFu + ((c.u >> 16) & 1u);   // round-to-nearest-even
    return (unsigned short)(u >> 16);
}

__device__ __forceinline__ v8f wmma_bf16(const BFrag& a, const BFrag& b, v8f c) {
    return __builtin_amdgcn_wmma_f32_16x16x32_bf16(
        false, a.v, false, b.v, (short)0, c, false, false);
}

// LDS byte offset of a generic pointer to __shared__ (low 32 bits of the
// LDS-aperture address are the LDS offset).
__device__ __forceinline__ unsigned lds_off(const void* p) {
    return (unsigned)(unsigned long long)p;
}

// GLOBAL_LOAD_ASYNC_TO_LDS_B128 (ASYNCcnt-tracked, bypasses VGPRs).
// saddr form: 64-bit uniform base in SGPRs + per-lane 32-bit byte offset.
__device__ __forceinline__ void async_g2l_b128(unsigned ldsAddr,
                                               unsigned long long gbase,
                                               unsigned gbyteOff) {
    asm volatile("global_load_async_to_lds_b128 %0, %1, %2"
                 :: "v"(ldsAddr), "v"(gbyteOff), "s"(gbase)
                 : "memory");
}

__device__ __forceinline__ void wait_async0() {
    asm volatile("s_wait_asynccnt 0x0" ::: "memory");
}

// ---------------------------------------------------------------------------
// f32 -> bf16 conversion (4-wide)
// ---------------------------------------------------------------------------
__global__ __launch_bounds__(256) void cvt4(const float* __restrict__ src,
                                            unsigned short* __restrict__ dst,
                                            int n4) {
    int i = blockIdx.x * 256 + threadIdx.x;
    if (i < n4) {
        const float4 v = reinterpret_cast<const float4*>(src)[i];
        ushort4 o;
        o.x = f2bf(v.x); o.y = f2bf(v.y); o.z = f2bf(v.z); o.w = f2bf(v.w);
        reinterpret_cast<ushort4*>(dst)[i] = o;
    }
}

// ---------------------------------------------------------------------------
// C[M,N] = X[M,K](bf16) @ W[N,K](bf16)^T + bias (+residual)
// 128x128x32 tiles; 256 threads = 8 waves (2M x 4N), 64x32 per wave.
// Double-buffered LDS fed by global_load_async_to_lds_b128.
// ---------------------------------------------------------------------------
#define GBM 128
#define GBN 128
#define GBK 32
#define GLDT 40   // padded LDS row stride (ushorts)

__global__ __launch_bounds__(256)
void gemm_bf16(const unsigned short* __restrict__ X,
               const unsigned short* __restrict__ W,
               const float* __restrict__ bias,
               const float* __restrict__ residual,   // nullable
               float* __restrict__ outF,             // nullable
               unsigned short* __restrict__ outB,    // nullable
               int M, int N, int K)
{
    __shared__ unsigned short sX[2][GBM * GLDT];
    __shared__ unsigned short sW[2][GBN * GLDT];

    const int tid  = threadIdx.x;
    const int lane = tid & 31;
    const int wave = tid >> 5;
    const int wm   = wave & 1;         // 0..1 (M)
    const int wn   = wave >> 1;        // 0..3 (N)
    const int half = lane >> 4;
    const int l16  = lane & 15;
    const int bm0  = blockIdx.y * GBM;
    const int bn0  = blockIdx.x * GBN;

    // ---- per-thread staging assignment: 2 chunks of 16B per tile
    const int cA = tid, cB = tid + 256;                 // 512 chunks / tile
    const int rA = cA >> 2, oA = (cA & 3) * 8;
    const int rB = cB >> 2, oB = (cB & 3) * 8;
    const unsigned long long Xull = (unsigned long long)X;
    const unsigned long long Wull = (unsigned long long)W;
    const unsigned gxA = (unsigned)(((size_t)(bm0 + rA) * K + oA) * 2);
    const unsigned gxB = (unsigned)(((size_t)(bm0 + rB) * K + oB) * 2);
    const unsigned gwA = (unsigned)(((size_t)(bn0 + rA) * K + oA) * 2);
    const unsigned gwB = (unsigned)(((size_t)(bn0 + rB) * K + oB) * 2);
    const unsigned lXA[2] = { lds_off(&sX[0][rA * GLDT + oA]),
                              lds_off(&sX[1][rA * GLDT + oA]) };
    const unsigned lXB[2] = { lds_off(&sX[0][rB * GLDT + oB]),
                              lds_off(&sX[1][rB * GLDT + oB]) };
    const unsigned lWA[2] = { lds_off(&sW[0][rA * GLDT + oA]),
                              lds_off(&sW[1][rA * GLDT + oA]) };
    const unsigned lWB[2] = { lds_off(&sW[0][rB * GLDT + oB]),
                              lds_off(&sW[1][rB * GLDT + oB]) };

    auto stage = [&](int k0, int buf) {
        const unsigned kb = (unsigned)(k0 * 2);
        async_g2l_b128(lXA[buf], Xull, gxA + kb);
        async_g2l_b128(lXB[buf], Xull, gxB + kb);
        async_g2l_b128(lWA[buf], Wull, gwA + kb);
        async_g2l_b128(lWB[buf], Wull, gwB + kb);
    };

    v8f acc[4][2];
    for (int mi = 0; mi < 4; ++mi)
        for (int ni = 0; ni < 2; ++ni)
            acc[mi][ni] = v8f{0.f,0.f,0.f,0.f,0.f,0.f,0.f,0.f};

    // ---- prologue: DMA tile 0
    stage(0, 0);
    wait_async0();
    __syncthreads();

    int buf = 0;
    for (int k0 = 0; k0 < K; k0 += GBK, buf ^= 1) {
        if (k0 + GBK < K) stage(k0 + GBK, buf ^ 1);   // DMA next tile (overlapped)

        const unsigned short* xb = sX[buf];
        const unsigned short* wb = sW[buf];

        // A fragments: lane = M row (l16); half-wave K split {h*8..+7, h*8+16..+23}
        BFrag a[4];
        #pragma unroll
        for (int mi = 0; mi < 4; ++mi) {
            const int m = wm * 64 + mi * 16 + l16;
            const unsigned short* p = &xb[m * GLDT + half * 8];
            a[mi].q[0] = *reinterpret_cast<const u32x4*>(p);
            a[mi].q[1] = *reinterpret_cast<const u32x4*>(p + 16);
        }
        // B fragments: lane = N col (l16); half-wave K split {h*8..+7, h*8+16..+23}
        BFrag bfr[2];
        #pragma unroll
        for (int ni = 0; ni < 2; ++ni) {
            const int nn = wn * 32 + ni * 16 + l16;
            const unsigned short* p = &wb[nn * GLDT + half * 8];
            bfr[ni].q[0] = *reinterpret_cast<const u32x4*>(p);
            bfr[ni].q[1] = *reinterpret_cast<const u32x4*>(p + 16);
        }
        #pragma unroll
        for (int mi = 0; mi < 4; ++mi)
            #pragma unroll
            for (int ni = 0; ni < 2; ++ni)
                acc[mi][ni] = wmma_bf16(a[mi], bfr[ni], acc[mi][ni]);

        wait_async0();        // next tile landed
        __syncthreads();      // publish to all waves / retire readers
    }

    // ---- epilogue: bias (+ residual), f32 and/or bf16 output
    for (int mi = 0; mi < 4; ++mi) {
        for (int ni = 0; ni < 2; ++ni) {
            const int col = bn0 + wn * 32 + ni * 16 + l16;
            const float bv = bias[col];
            #pragma unroll
            for (int i = 0; i < 8; ++i) {
                const int row = bm0 + wm * 64 + mi * 16 + half * 8 + i;
                float v = acc[mi][ni][i] + bv;
                if (residual) v += residual[(size_t)row * N + col];
                if (outF) outF[(size_t)row * N + col] = v;
                if (outB) outB[(size_t)row * N + col] = f2bf(v);
            }
        }
    }
}

// ---------------------------------------------------------------------------
// Flash attention: one WG per (128-query tile, head, batch); 8 waves,
// 16 query rows/wave.  32-key K/V tiles streamed through LDS (K tile via
// async DMA, V staged transposed), online softmax in f32.
// ---------------------------------------------------------------------------
#define AKT 32

__global__ __launch_bounds__(256)
void attn(const unsigned short* __restrict__ Qb,
          const unsigned short* __restrict__ Kb,
          const unsigned short* __restrict__ Vb,
          unsigned short* __restrict__ ctxb,
          int T)
{
    __shared__ unsigned short sK[AKT * 72];        // [key][d], padded
    __shared__ unsigned short sV[64 * 40];         // transposed [d][key], padded
    __shared__ unsigned short sP[8 * 16 * 40];     // per-wave P scratch

    const int Hn   = 1024;
    const int tid  = threadIdx.x;
    const int lane = tid & 31;
    const int wave = tid >> 5;
    const int half = lane >> 4;
    const int l16  = lane & 15;
    const int qt   = blockIdx.x;
    const int head = blockIdx.y;
    const int b    = blockIdx.z;

    // ---- async K-tile staging: 256 chunks of 16B, one per thread
    const int kKey = tid >> 3, kcc = (tid & 7) * 8;
    const unsigned long long Kull = (unsigned long long)Kb;
    const unsigned gK0 =
        (unsigned)(((size_t)(b * T + kKey) * Hn + head * 64 + kcc) * 2);
    const unsigned lKd = lds_off(&sK[kKey * 72 + kcc]);

    // ---- preload this wave's Q fragments (16 rows x 64 d = 2 K-steps)
    BFrag aq[2];
    {
        const int row = b * T + qt * 128 + wave * 16 + l16;
        const unsigned short* qp = &Qb[(size_t)row * Hn + head * 64];
        #pragma unroll
        for (int s = 0; s < 2; ++s) {
            const unsigned short* p = qp + s * 32 + half * 8;
            aq[s].q[0] = *reinterpret_cast<const u32x4*>(p);
            aq[s].q[1] = *reinterpret_cast<const u32x4*>(p + 16);
        }
    }

    v8f oacc[4];
    for (int jd = 0; jd < 4; ++jd)
        oacc[jd] = v8f{0.f,0.f,0.f,0.f,0.f,0.f,0.f,0.f};
    float mrow[8], lrow[8];
    for (int i = 0; i < 8; ++i) { mrow[i] = -3.0e38f; lrow[i] = 0.f; }

    const int nkt = T / AKT;
    for (int kt = 0; kt < nkt; ++kt) {
        __syncthreads();   // previous iteration's readers are done
        // ---- K tile: one async b128 per thread
        async_g2l_b128(lKd, Kull, gK0 + (unsigned)(kt * AKT * Hn * 2));
        // ---- V tile transposed: sV[d][key]
        #pragma unroll
        for (int e = tid; e < AKT * 64; e += 256) {
            const int key = e >> 6, d = e & 63;
            sV[d * 40 + key] =
                Vb[(size_t)(b * T + kt * AKT + key) * Hn + head * 64 + d];
        }
        wait_async0();
        __syncthreads();

        // ---- S = Q K^T : 2 key-subtiles x 2 d-ksteps
        v8f sa[2];
        sa[0] = v8f{0.f,0.f,0.f,0.f,0.f,0.f,0.f,0.f};
        sa[1] = v8f{0.f,0.f,0.f,0.f,0.f,0.f,0.f,0.f};
        #pragma unroll
        for (int j = 0; j < 2; ++j) {
            #pragma unroll
            for (int s = 0; s < 2; ++s) {
                BFrag bk;
                const unsigned short* kp = &sK[(j * 16 + l16) * 72 + s * 32 + half * 8];
                bk.q[0] = *reinterpret_cast<const u32x4*>(kp);
                bk.q[1] = *reinterpret_cast<const u32x4*>(kp + 16);
                sa[j] = wmma_bf16(aq[s], bk, sa[j]);
            }
        }

        // ---- online softmax (row r = i + half*8; stats replicated per 16-lane half)
        float pn0[8], pn1[8];
        #pragma unroll
        for (int i = 0; i < 8; ++i) {
            const float s0 = sa[0][i] * 0.125f;
            const float s1 = sa[1][i] * 0.125f;
            float mx = fmaxf(s0, s1);
            for (int d = 1; d < 16; d <<= 1) mx = fmaxf(mx, __shfl_xor(mx, d, 32));
            const float mnew = fmaxf(mrow[i], mx);
            const float corr = __expf(mrow[i] - mnew);
            const float p0 = __expf(s0 - mnew);
            const float p1 = __expf(s1 - mnew);
            float rs = p0 + p1;
            for (int d = 1; d < 16; d <<= 1) rs += __shfl_xor(rs, d, 32);
            lrow[i] = lrow[i] * corr + rs;
            mrow[i] = mnew;
            for (int jd = 0; jd < 4; ++jd) oacc[jd][i] *= corr;
            pn0[i] = p0; pn1[i] = p1;
        }

        // ---- re-swizzle P: C-layout -> A-layout via per-wave LDS scratch
        unsigned short* Pw = &sP[wave * 16 * 40];
        #pragma unroll
        for (int i = 0; i < 8; ++i) {
            const int r = i + half * 8;
            Pw[r * 40 + l16]      = f2bf(pn0[i]);
            Pw[r * 40 + 16 + l16] = f2bf(pn1[i]);
        }
        asm volatile("s_wait_dscnt 0x0" ::: "memory");   // wave-internal LDS RAW
        BFrag pa;
        {
            const unsigned short* pp = &Pw[l16 * 40 + half * 8];
            pa.q[0] = *reinterpret_cast<const u32x4*>(pp);
            pa.q[1] = *reinterpret_cast<const u32x4*>(pp + 16);
        }

        // ---- O += P V : 4 d-subtiles, contraction over 32 keys
        #pragma unroll
        for (int jd = 0; jd < 4; ++jd) {
            BFrag bv;
            const unsigned short* vp = &sV[(jd * 16 + l16) * 40 + half * 8];
            bv.q[0] = *reinterpret_cast<const u32x4*>(vp);
            bv.q[1] = *reinterpret_cast<const u32x4*>(vp + 16);
            oacc[jd] = wmma_bf16(pa, bv, oacc[jd]);
        }
    }

    // ---- normalize + store ctx (bf16) for the O-projection GEMM
    for (int jd = 0; jd < 4; ++jd) {
        #pragma unroll
        for (int i = 0; i < 8; ++i) {
            const int row = b * T + qt * 128 + wave * 16 + half * 8 + i;
            const int col = head * 64 + jd * 16 + l16;
            ctxb[(size_t)row * Hn + col] = f2bf(oacc[jd][i] / lrow[i]);
        }
    }
}

// ---------------------------------------------------------------------------
// LayerNorm over H=1024, one workgroup (256 threads) per row.
// ---------------------------------------------------------------------------
__global__ __launch_bounds__(256)
void layernorm(const float* __restrict__ x,
               const float* __restrict__ gamma,
               const float* __restrict__ beta,
               float* __restrict__ out, int Hn)
{
    __shared__ float rs[8], rs2[8];
    const int row = blockIdx.x;
    const int tid = threadIdx.x;
    const float* xr = x + (size_t)row * Hn;

    float v[4], s = 0.f, s2 = 0.f;
    #pragma unroll
    for (int i = 0; i < 4; ++i) {
        v[i] = xr[tid + i * 256];
        s += v[i]; s2 += v[i] * v[i];
    }
    for (int d = 1; d < 32; d <<= 1) {
        s  += __shfl_xor(s, d, 32);
        s2 += __shfl_xor(s2, d, 32);
    }
    if ((tid & 31) == 0) { rs[tid >> 5] = s; rs2[tid >> 5] = s2; }
    __syncthreads();
    float ts = 0.f, ts2 = 0.f;
    for (int w = 0; w < 8; ++w) { ts += rs[w]; ts2 += rs2[w]; }
    const float mu  = ts * (1.0f / 1024.0f);
    const float var = ts2 * (1.0f / 1024.0f) - mu * mu;
    const float inv = rsqrtf(var + 1e-6f);
    #pragma unroll
    for (int i = 0; i < 4; ++i) {
        const int col = tid + i * 256;
        out[(size_t)row * Hn + col] = (v[i] - mu) * inv * gamma[col] + beta[col];
    }
}

// ---------------------------------------------------------------------------
// Host-side launch
// ---------------------------------------------------------------------------
extern "C" void kernel_launch(void* const* d_in, const int* in_sizes, int n_in,
                              void* d_out, int out_size, void* d_ws, size_t ws_size,
                              hipStream_t stream) {
    constexpr int B = 2, T = 2048, Hn = 1024;
    constexpr int M = B * T;                       // 4096 rows

    const float* h     = (const float*)d_in[0];
    const float* Wq    = (const float*)d_in[1];
    const float* bq    = (const float*)d_in[2];
    const float* Wk    = (const float*)d_in[3];
    const float* bk    = (const float*)d_in[4];
    const float* Wv    = (const float*)d_in[5];
    const float* bv    = (const float*)d_in[6];
    const float* Wo    = (const float*)d_in[7];
    const float* bo    = (const float*)d_in[8];
    const float* gamma = (const float*)d_in[9];
    const float* beta  = (const float*)d_in[10];

    size_t off = 0;
    auto carve = [&](size_t bytes) {
        void* p = (char*)d_ws + off;
        off += (bytes + 255) & ~(size_t)255;
        return p;
    };
    unsigned short* hb    = (unsigned short*)carve((size_t)M * Hn * 2);
    unsigned short* Wqb   = (unsigned short*)carve((size_t)Hn * Hn * 2);
    unsigned short* Wkb   = (unsigned short*)carve((size_t)Hn * Hn * 2);
    unsigned short* Wvb   = (unsigned short*)carve((size_t)Hn * Hn * 2);
    unsigned short* Wob   = (unsigned short*)carve((size_t)Hn * Hn * 2);
    unsigned short* Qb    = (unsigned short*)carve((size_t)M * Hn * 2);
    unsigned short* Kb    = (unsigned short*)carve((size_t)M * Hn * 2);
    unsigned short* Vb    = (unsigned short*)carve((size_t)M * Hn * 2);
    unsigned short* ctxb  = (unsigned short*)carve((size_t)M * Hn * 2);
    float*          preLN = (float*)carve((size_t)M * Hn * 4);

    {
        const int n4h = (M * Hn) / 4, n4w = (Hn * Hn) / 4;
        cvt4<<<(n4h + 255) / 256, 256, 0, stream>>>(h,  hb,  n4h);
        cvt4<<<(n4w + 255) / 256, 256, 0, stream>>>(Wq, Wqb, n4w);
        cvt4<<<(n4w + 255) / 256, 256, 0, stream>>>(Wk, Wkb, n4w);
        cvt4<<<(n4w + 255) / 256, 256, 0, stream>>>(Wv, Wvb, n4w);
        cvt4<<<(n4w + 255) / 256, 256, 0, stream>>>(Wo, Wob, n4w);
    }

    dim3 ggrid(Hn / GBN, M / GBM);   // 8 x 32
    gemm_bf16<<<ggrid, 256, 0, stream>>>(hb, Wqb, bq, nullptr, nullptr, Qb, M, Hn, Hn);
    gemm_bf16<<<ggrid, 256, 0, stream>>>(hb, Wkb, bk, nullptr, nullptr, Kb, M, Hn, Hn);
    gemm_bf16<<<ggrid, 256, 0, stream>>>(hb, Wvb, bv, nullptr, nullptr, Vb, M, Hn, Hn);

    attn<<<dim3(T / 128, 16, B), 256, 0, stream>>>(Qb, Kb, Vb, ctxb, T);

    gemm_bf16<<<ggrid, 256, 0, stream>>>(ctxb, Wob, bo, h, preLN, nullptr, M, Hn, Hn);

    layernorm<<<M, 256, 0, stream>>>(preLN, gamma, beta, (float*)d_out, Hn);

    (void)in_sizes; (void)n_in; (void)out_size; (void)ws_size;
}
